// Attention_23880018166059
// MI455X (gfx1250) — matrix operand order repaired and checked by reading, not compiled, not run on hardware
//
#include <hip/hip_runtime.h>
#include <math.h>

typedef __attribute__((ext_vector_type(16))) _Float16 v16h;
typedef __attribute__((ext_vector_type(8)))  _Float16 v8h;
typedef __attribute__((ext_vector_type(2)))  _Float16 v2h;
typedef __attribute__((ext_vector_type(8)))  float    v8f;

#define SEQ_LEN 4096
#define DMODEL  1024
#define NHEADS  16
#define DHEAD   64

// ---------------- WMMA fragment helpers (CDNA5 16x16x32 f16 layouts) ----------------

// A-matrix 16x32 f16: lane L -> row m=L&15, half h=L>>4.
// Per lane: two contiguous 8-half chunks at k0+8h and k0+16+8h.
static __device__ __forceinline__ v16h ld_afrag(const _Float16* base, int lda, int lane) {
    const int m  = lane & 15;
    const int hf = lane >> 4;
    const _Float16* p = base + (size_t)m * lda + 8 * hf;
    v8h lo = *(const v8h*)(p);
    v8h hi = *(const v8h*)(p + 16);
    v16h a;
#pragma unroll
    for (int i = 0; i < 8; ++i) { a[i] = lo[i]; a[i + 8] = hi[i]; }
    return a;
}

// B-matrix 32x16 f16 loaded from B^T stored row-major (rows = output column n).
// lane L -> col n=L&15, half h=L>>4; two 8-half chunks at k0+8h and k0+16+8h.
static __device__ __forceinline__ v16h ld_bfrag(const _Float16* baseT, int ldb, int lane) {
    const int n  = lane & 15;
    const int kb = (lane >> 4) << 3;
    const _Float16* p = baseT + (size_t)n * ldb + kb;
    v8h lo = *(const v8h*)(p);
    v8h hi = *(const v8h*)(p + 16);
    v16h b;
#pragma unroll
    for (int i = 0; i < 8; ++i) { b[i] = lo[i]; b[i + 8] = hi[i]; }
    return b;
}

static __device__ __forceinline__ v8f wmma16(v16h a, v16h b, v8f c) {
    return __builtin_amdgcn_wmma_f32_16x16x32_f16(false, a, false, b, (short)0, c, false, false);
}

// ---------------- Kernel 1: f32 -> f16 convert ----------------

__global__ void cvt_f32_to_f16(const float* __restrict__ src, _Float16* __restrict__ dst, int n) {
    int i = blockIdx.x * blockDim.x + threadIdx.x;
    int stride = gridDim.x * blockDim.x;
    for (; i < n; i += stride) dst[i] = (_Float16)src[i];
}

// ---------------- shared GEMM tile stager (64x64 f16 -> LDS stride 72) ----------------
static __device__ __forceinline__ void stage_wtile(_Float16* dst, const _Float16* src,
                                                   int ldsrc, int tid) {
    int r   = tid >> 1;
    int seg = (tid & 1) * 32;
    const _Float16* gp = src + (size_t)r * ldsrc + seg;
#pragma unroll
    for (int j = 0; j < 4; ++j)
        *(v8h*)&dst[r * 72 + seg + j * 8] = *(const v8h*)(gp + j * 8);
    __builtin_prefetch(gp + 64, 0, 1);  // speculative: next-next k tile
}

// ---------------- Kernel 2: fused QKV projection ----------------
// 128 threads = 4 waves; each wave owns a 32x64 output tile (8 accumulators).
// Double-buffered LDS weight tiles: one barrier per k-step, staging overlaps WMMA.
// wh:  [48][64][1024] f16  (0-15: Q heads, 16-31: K heads, 32-47: V heads)
// qkv: [48][4096][64] f16
__global__ void __launch_bounds__(128)
qkv_proj(const _Float16* __restrict__ xh, const _Float16* __restrict__ wh,
         _Float16* __restrict__ qkv) {
    __shared__ _Float16 wtile[2][64 * 72];
    const int mh  = blockIdx.y;  // 0..47
    const _Float16* w  = wh  + (size_t)mh * 64 * DMODEL;
    _Float16*       ob = qkv + (size_t)mh * SEQ_LEN * DHEAD;
    const int wave = threadIdx.x >> 5;
    const int lane = threadIdx.x & 31;
    const int row0 = blockIdx.x * 128 + wave * 32;

    v8f acc[2][4] = {};
    stage_wtile(wtile[0], w, DMODEL, threadIdx.x);
    __syncthreads();

    int buf = 0;
    for (int k0 = 0; k0 < DMODEL; k0 += 64, buf ^= 1) {
        if (k0 + 64 < DMODEL)
            stage_wtile(wtile[buf ^ 1], w + k0 + 64, DMODEL, threadIdx.x);
#pragma unroll
        for (int ks = 0; ks < 64; ks += 32) {
            v16h b[4];
#pragma unroll
            for (int t = 0; t < 4; ++t)
                b[t] = ld_bfrag(&wtile[buf][(t * 16) * 72 + ks], 72, lane);
            v16h a[2];
            a[0] = ld_afrag(xh + (size_t)row0 * DMODEL + k0 + ks, DMODEL, lane);
            a[1] = ld_afrag(xh + (size_t)(row0 + 16) * DMODEL + k0 + ks, DMODEL, lane);
            // mi-outer order keeps all 4 B frags live -> one DS wait, 8 WMMAs
#pragma unroll
            for (int mi = 0; mi < 2; ++mi)
#pragma unroll
                for (int t = 0; t < 4; ++t)
                    acc[mi][t] = wmma16(a[mi], b[t], acc[mi][t]);
        }
        __syncthreads();
    }
    const int n  = lane & 15;
    const int hf = lane >> 4;
#pragma unroll
    for (int mi = 0; mi < 2; ++mi)
#pragma unroll
        for (int t = 0; t < 4; ++t)
#pragma unroll
            for (int r = 0; r < 8; ++r)
                ob[(size_t)(row0 + mi * 16 + r + 8 * hf) * DHEAD + t * 16 + n] =
                    (_Float16)acc[mi][t][r];
}

// ---------------- Kernel 3: causal flash attention ----------------
// One block = 128 threads (4 waves), one (head, 64-query block); 16 q-rows/wave.
__global__ void __launch_bounds__(128)
flash_attn(const _Float16* __restrict__ qkv, _Float16* __restrict__ z) {
    __shared__ _Float16 Ktile[64 * 72];     // K rows [key][d]
    __shared__ _Float16 Vtile[64 * 72];     // V^T    [d][key]
    __shared__ _Float16 Ptile[4][16 * 72];  // per-wave P [qrow][key]

    const int h     = blockIdx.y;
    const int qb    = blockIdx.x;
    const int wave  = threadIdx.x >> 5;
    const int lane  = threadIdx.x & 31;
    const int qrow0 = qb * 64 + wave * 16;

    const _Float16* Qb = qkv + ((size_t)(0 * NHEADS + h) * SEQ_LEN + qrow0) * DHEAD;
    const _Float16* Kb = qkv + ((size_t)(1 * NHEADS + h) * SEQ_LEN) * DHEAD;
    const _Float16* Vb = qkv + ((size_t)(2 * NHEADS + h) * SEQ_LEN) * DHEAD;

    v16h qf[2];
    qf[0] = ld_afrag(Qb, DHEAD, lane);
    qf[1] = ld_afrag(Qb + 32, DHEAD, lane);

    v8f accO[4] = {};
    float rowM[8], rowS[8];
#pragma unroll
    for (int r = 0; r < 8; ++r) { rowM[r] = -3.0e38f; rowS[r] = 0.0f; }

    const int n  = lane & 15;
    const int hf = lane >> 4;

    for (int kt = 0; kt <= qb; ++kt) {
        const int k0g = kt * 64;
        {   // stage K tile (row-major copy, vector b128)
            int key = threadIdx.x >> 1;
            int seg = (threadIdx.x & 1) * 32;
            const _Float16* kp = Kb + (size_t)(k0g + key) * DHEAD + seg;
#pragma unroll
            for (int j = 0; j < 4; ++j)
                *(v8h*)&Ktile[key * 72 + seg + j * 8] = *(const v8h*)(kp + j * 8);
        }
        {   // stage V^T tile: key-pair transpose, packed 32-bit LDS stores
            int pair = threadIdx.x & 31;          // keys 2p, 2p+1
            int dseg = (threadIdx.x >> 5) * 16;   // 16 d-values per thread
            const _Float16* vp0 = Vb + (size_t)(k0g + 2 * pair) * DHEAD + dseg;
            const _Float16* vp1 = vp0 + DHEAD;
            v8h v0a = *(const v8h*)(vp0);
            v8h v0b = *(const v8h*)(vp0 + 8);
            v8h v1a = *(const v8h*)(vp1);
            v8h v1b = *(const v8h*)(vp1 + 8);
#pragma unroll
            for (int e = 0; e < 8; ++e) {
                v2h pa; pa[0] = v0a[e]; pa[1] = v1a[e];
                *(v2h*)&Vtile[(dseg + e) * 72 + 2 * pair] = pa;
                v2h pb; pb[0] = v0b[e]; pb[1] = v1b[e];
                *(v2h*)&Vtile[(dseg + 8 + e) * 72 + 2 * pair] = pb;
            }
        }
        __syncthreads();

        // S = Q K^T (16q x 64k): batch 4 B-frags per d-step, then 4 WMMAs
        v8f sc[4] = {};
#pragma unroll
        for (int ds = 0; ds < 2; ++ds) {
            v16h kb[4];
#pragma unroll
            for (int t = 0; t < 4; ++t)
                kb[t] = ld_bfrag(&Ktile[(t * 16) * 72 + ds * 32], 72, lane);
#pragma unroll
            for (int t = 0; t < 4; ++t)
                sc[t] = wmma16(qf[ds], kb[t], sc[t]);
        }

        // scale + causal mask (C layout: row m=r+8*hf, col = 16t+n)
        const bool diag = (kt == qb);
#pragma unroll
        for (int t = 0; t < 4; ++t)
#pragma unroll
            for (int r = 0; r < 8; ++r) {
                float s = sc[t][r] * 0.125f;  // 1/sqrt(64)
                if (diag && (k0g + t * 16 + n) > (qrow0 + r + 8 * hf)) s = -3.0e38f;
                sc[t][r] = s;
            }

        // online softmax per row
#pragma unroll
        for (int r = 0; r < 8; ++r) {
            float m = fmaxf(fmaxf(sc[0][r], sc[1][r]), fmaxf(sc[2][r], sc[3][r]));
#pragma unroll
            for (int off = 1; off < 16; off <<= 1)
                m = fmaxf(m, __shfl_xor(m, off, 32));
            float newM  = fmaxf(rowM[r], m);
            float alpha = __expf(rowM[r] - newM);
            float psum  = 0.0f;
#pragma unroll
            for (int t = 0; t < 4; ++t) {
                float p = __expf(sc[t][r] - newM);
                sc[t][r] = p;
                psum += p;
                accO[t][r] *= alpha;
            }
#pragma unroll
            for (int off = 1; off < 16; off <<= 1)
                psum += __shfl_xor(psum, off, 32);
            rowS[r] = rowS[r] * alpha + psum;
            rowM[r] = newM;
#pragma unroll
            for (int t = 0; t < 4; ++t)  // P -> per-wave LDS (re-layout to A-frag)
                Ptile[wave][(r + 8 * hf) * 72 + t * 16 + n] = (_Float16)sc[t][r];
        }
        // wave-local ordering: P writes must land before A-frag reads
        asm volatile("s_wait_dscnt 0" ::: "memory");

        // O += P V: batch A-frag + 4 B-frags per k-step, then 4 WMMAs
#pragma unroll
        for (int ks = 0; ks < 2; ++ks) {
            v16h a = ld_afrag(&Ptile[wave][ks * 32], 72, lane);
            v16h vb[4];
#pragma unroll
            for (int t = 0; t < 4; ++t)
                vb[t] = ld_bfrag(&Vtile[(t * 16) * 72 + ks * 32], 72, lane);
#pragma unroll
            for (int t = 0; t < 4; ++t)
                accO[t] = wmma16(a, vb[t], accO[t]);
        }
        __syncthreads();
    }

    // normalize and write z[q][h*64 + d]
#pragma unroll
    for (int r = 0; r < 8; ++r) {
        float inv = 1.0f / rowS[r];
        int q = qrow0 + r + 8 * hf;
#pragma unroll
        for (int t = 0; t < 4; ++t)
            z[(size_t)q * DMODEL + h * DHEAD + t * 16 + n] = (_Float16)(accO[t][r] * inv);
    }
}

// ---------------- Kernel 4: output projection (f32 out) ----------------
// 128 threads = 4 waves; each wave owns a 32x64 output tile; double-buffered LDS.
__global__ void __launch_bounds__(128)
out_proj(const _Float16* __restrict__ zh, const _Float16* __restrict__ woh,
         float* __restrict__ out) {
    __shared__ _Float16 wtile[2][64 * 72];
    const int wave = threadIdx.x >> 5;
    const int lane = threadIdx.x & 31;
    const int row0 = blockIdx.x * 128 + wave * 32;
    const int n0g  = blockIdx.y * 64;
    const _Float16* w = woh + (size_t)n0g * DMODEL;

    v8f acc[2][4] = {};
    stage_wtile(wtile[0], w, DMODEL, threadIdx.x);
    __syncthreads();

    int buf = 0;
    for (int k0 = 0; k0 < DMODEL; k0 += 64, buf ^= 1) {
        if (k0 + 64 < DMODEL)
            stage_wtile(wtile[buf ^ 1], w + k0 + 64, DMODEL, threadIdx.x);
#pragma unroll
        for (int ks = 0; ks < 64; ks += 32) {
            v16h b[4];
#pragma unroll
            for (int t = 0; t < 4; ++t)
                b[t] = ld_bfrag(&wtile[buf][(t * 16) * 72 + ks], 72, lane);
            v16h a[2];
            a[0] = ld_afrag(zh + (size_t)row0 * DMODEL + k0 + ks, DMODEL, lane);
            a[1] = ld_afrag(zh + (size_t)(row0 + 16) * DMODEL + k0 + ks, DMODEL, lane);
#pragma unroll
            for (int mi = 0; mi < 2; ++mi)
#pragma unroll
                for (int t = 0; t < 4; ++t)
                    acc[mi][t] = wmma16(a[mi], b[t], acc[mi][t]);
        }
        __syncthreads();
    }
    const int n  = lane & 15;
    const int hf = lane >> 4;
#pragma unroll
    for (int mi = 0; mi < 2; ++mi)
#pragma unroll
        for (int t = 0; t < 4; ++t)
#pragma unroll
            for (int r = 0; r < 8; ++r)
                out[(size_t)(row0 + mi * 16 + r + 8 * hf) * DMODEL + n0g + t * 16 + n] =
                    acc[mi][t][r];
}

// ---------------- Host launcher ----------------
extern "C" void kernel_launch(void* const* d_in, const int* in_sizes, int n_in,
                              void* d_out, int out_size, void* d_ws, size_t ws_size,
                              hipStream_t stream) {
    (void)in_sizes; (void)n_in; (void)out_size; (void)ws_size;
    const float* x  = (const float*)d_in[0];
    const float* wk = (const float*)d_in[1];
    const float* wq = (const float*)d_in[2];
    const float* wv = (const float*)d_in[3];
    const float* wo = (const float*)d_in[4];
    float* out = (float*)d_out;

    // workspace layout (f16 buffers), 48 MiB total
    char* ws = (char*)d_ws;
    _Float16* xh  = (_Float16*)(ws);                    //  8 MiB  x as f16        [4096][1024]
    _Float16* wh  = (_Float16*)(ws + 8388608);          //  6 MiB  QKV weights f16 [48][64][1024]
    _Float16* woh = (_Float16*)(ws + 14680064);         //  2 MiB  W_O f16         [1024][1024]
    _Float16* qkv = (_Float16*)(ws + 16777216);         // 24 MiB  Q/K/V           [48][4096][64]
    _Float16* zh  = (_Float16*)(ws + 41943040);         //  8 MiB  z               [4096][1024]

    const int HW = NHEADS * DHEAD * DMODEL;  // 1,048,576 elements per projection
    cvt_f32_to_f16<<<512, 256, 0, stream>>>(x,  xh,           SEQ_LEN * DMODEL);
    cvt_f32_to_f16<<<256, 256, 0, stream>>>(wq, wh,           HW);
    cvt_f32_to_f16<<<256, 256, 0, stream>>>(wk, wh + HW,      HW);
    cvt_f32_to_f16<<<256, 256, 0, stream>>>(wv, wh + 2 * HW,  HW);
    cvt_f32_to_f16<<<256, 256, 0, stream>>>(wo, woh,          DMODEL * DMODEL);

    qkv_proj  <<<dim3(SEQ_LEN / 128, 48), 128, 0, stream>>>(xh, wh, qkv);
    flash_attn<<<dim3(SEQ_LEN / 64, NHEADS), 128, 0, stream>>>(qkv, zh);
    out_proj  <<<dim3(SEQ_LEN / 128, DMODEL / 64), 128, 0, stream>>>(zh, woh, out);
}
